// Agg2D_45827301048340
// MI455X (gfx1250) — matrix operand order repaired and checked by reading, not compiled, not run on hardware
//
#include <hip/hip_runtime.h>
#include <hip/hip_bf16.h>
#include <math.h>

// ---------------------------------------------------------------------------
// Types for CDNA5 WMMA (wave32, 16x16x32 f16 -> f32)
// ---------------------------------------------------------------------------
typedef _Float16 h4v  __attribute__((ext_vector_type(4)));
typedef _Float16 h8v  __attribute__((ext_vector_type(8)));
typedef _Float16 v16h __attribute__((ext_vector_type(16)));
typedef float    v8f  __attribute__((ext_vector_type(8)));

__device__ __forceinline__ float gelu_exact(float x) {
  // jax.nn.gelu(approximate=False): 0.5*x*(1+erf(x/sqrt(2)))
  return 0.5f * x * (1.0f + erff(x * 0.70710678118654752440f));
}

// ---------------------------------------------------------------------------
// Zero-fill (graph-capture safe)
// ---------------------------------------------------------------------------
__global__ void zero_kernel(float* __restrict__ p, size_t n) {
  size_t i = (size_t)blockIdx.x * blockDim.x + threadIdx.x;
  if (i < n) p[i] = 0.0f;
}

// ---------------------------------------------------------------------------
// Weight pre-pack: W (K x N f32, row-major) -> Bp (Np x Kp f16, N-major,
// zero padded, Kp%32==0, Np%64==0). Makes GEMM B-staging a pure 16B/lane
// DMA copy with no guards and no conversion.
// ---------------------------------------------------------------------------
__global__ void pack_b_f16(const float* __restrict__ W, _Float16* __restrict__ out,
                           int K, int N, int Kp, int Np) {
  size_t tot = (size_t)Np * Kp;
  size_t i = (size_t)blockIdx.x * blockDim.x + threadIdx.x;
  if (i >= tot) return;
  int k = (int)(i % Kp);
  int n = (int)(i / Kp);
  float v = (k < K && n < N) ? W[(size_t)k * N + n] : 0.0f;
  out[i] = (_Float16)v;
}

// Pad MLP input (16384 x 12) -> (16384 x 32), zero padded
__global__ void pad_in_kernel(const float* __restrict__ in, float* __restrict__ out) {
  size_t i = (size_t)blockIdx.x * blockDim.x + threadIdx.x;
  if (i >= (size_t)16384 * 32) return;
  int k = (int)(i & 31);
  size_t m = i >> 5;
  out[i] = (k < 12) ? in[m * 12 + k] : 0.0f;
}

// ---------------------------------------------------------------------------
// Tiled WMMA GEMM:  C[M,N] = EPI(A[M,Kp] * B[Kp,N] + bias)
// Requirements: M % 128 == 0, Kp % 32 == 0 (A stride = Kp), Bp is packed
// (ceil64(N) x Kp, f16, N-major). Output stride = ldc (>= N).
//   EPI 0: acc + bias
//   EPI 1: gelu(acc + bias)
//   EPI 2: res + (acc + bias) * gamma     (ConvNeXt layer-scale residual)
// Block: 256 threads = 8 wave32 waves; tile 128(M) x 64(N) x 32(K).
// B tile is staged with CDNA5 async global->LDS DMA (ASYNCcnt);
// A tile via b128 loads + cvt; fragments read as contiguous 16B ds loads.
// ---------------------------------------------------------------------------
template <int EPI>
__global__ __launch_bounds__(256) void gemm_wmma(
    const float* __restrict__ A, const _Float16* __restrict__ Bp,
    const float* __restrict__ bias, float* __restrict__ Cc,
    int M, int N, int Kp, int ldc,
    const float* __restrict__ gamma, const float* __restrict__ res) {
  __shared__ __align__(16) _Float16 As[128][40];
  __shared__ __align__(16) _Float16 BsT[64][40];

  const int tid  = threadIdx.x;
  const int lane = tid & 31;
  const int wave = tid >> 5;
  const int wm   = wave >> 1;  // 0..3 -> 32-row band
  const int wn   = wave & 1;   // 0..1 -> 32-col band
  const int bm   = blockIdx.y * 128;
  const int bn   = blockIdx.x * 64;

  const int kABase = (lane >= 16) ? 8 : 0;
  const int kBBase = (lane >= 16) ? 8 : 0;
  const int mh     = (lane >= 16) ? 8 : 0;

  const int brow = tid >> 2;        // 0..63 : BsT row (N-local)
  const int bseg = (tid & 3) * 8;   // half index 0,8,16,24

  v8f acc[2][2] = {};

  for (int k0 = 0; k0 < Kp; k0 += 32) {
    // ---- B tile: async DMA, 16 bytes per lane, guard-free ----
    {
      const _Float16* src = Bp + (size_t)(bn + brow) * Kp + (k0 + bseg);
      unsigned lds = (unsigned)(unsigned long long)(void*)&BsT[brow][bseg];
      unsigned long long ga = (unsigned long long)src;
      asm volatile("global_load_async_to_lds_b128 %0, %1, off"
                   :: "v"(lds), "v"(ga)
                   : "memory");
    }
    // ---- A tile: 4x float4 loads, cvt f32->f16, 8B ds stores ----
#pragma unroll
    for (int i = 0; i < 4; ++i) {
      int v  = i * 256 + tid;     // 0..1023 over 128x8 float4s
      int r  = v >> 3;            // row 0..127
      int cb = (v & 7) * 4;       // col 0,4,...,28
      const float4 f = *(const float4*)(A + (size_t)(bm + r) * Kp + (k0 + cb));
      h4v hv = {(_Float16)f.x, (_Float16)f.y, (_Float16)f.z, (_Float16)f.w};
      *(h4v*)&As[r][cb] = hv;
    }
    if (k0 + 32 < Kp) {  // emits global_prefetch_b8
      __builtin_prefetch(A + (size_t)(bm + (tid >> 1)) * Kp + (k0 + 32), 0, 1);
    }
    asm volatile("s_wait_asynccnt 0" ::: "memory");
    __syncthreads();

#pragma unroll
    for (int tm = 0; tm < 2; ++tm) {
      const int row = wm * 32 + tm * 16 + (lane & 15);
      h8v alo = *(const h8v*)&As[row][kABase];
      h8v ahi = *(const h8v*)&As[row][kABase + 16];
      v16h a;
#pragma unroll
      for (int i = 0; i < 8; ++i) { a[i] = alo[i]; a[8 + i] = ahi[i]; }
#pragma unroll
      for (int tn = 0; tn < 2; ++tn) {
        const int col = wn * 32 + tn * 16 + (lane & 15);
        h8v blo = *(const h8v*)&BsT[col][kBBase];
        h8v bhi = *(const h8v*)&BsT[col][kBBase + 16];
        v16h b;
#pragma unroll
        for (int i = 0; i < 8; ++i) { b[i] = blo[i]; b[8 + i] = bhi[i]; }
        acc[tm][tn] = __builtin_amdgcn_wmma_f32_16x16x32_f16(
            false, a, false, b, (short)0, acc[tm][tn], false, false);
      }
    }
    __syncthreads();
  }

  // ---- epilogue (M always tile-exact; only N needs a store guard) ----
#pragma unroll
  for (int tm = 0; tm < 2; ++tm) {
#pragma unroll
    for (int tn = 0; tn < 2; ++tn) {
      const int gn = bn + wn * 32 + tn * 16 + (lane & 15);
      if (gn >= N) continue;
      const float bv = bias[gn];
      const float gv = (EPI == 2) ? gamma[gn] : 0.0f;
#pragma unroll
      for (int r = 0; r < 8; ++r) {
        const int gm = bm + wm * 32 + tm * 16 + r + mh;
        float v = acc[tm][tn][r] + bv;
        if (EPI == 1) v = gelu_exact(v);
        if (EPI == 2) v = res[(size_t)gm * ldc + gn] + v * gv;
        Cc[(size_t)gm * ldc + gn] = v;
      }
    }
  }
}

// ---------------------------------------------------------------------------
// Scatter-add of point features into the (4,256,256,32) grid.
// feat has padded row stride 1344.
// ---------------------------------------------------------------------------
__global__ void scatter_kernel(const float* __restrict__ inputs,
                               const float* __restrict__ feat,
                               float* __restrict__ grid) {
  const int P = 4096, TOT = 1312, FSTR = 1344;
  const int pid = blockIdx.x;  // b*4096 + p
  const int b   = pid / P;
  const float* in = inputs + (size_t)pid * 12;
  const int p0y = (int)((in[1] * 0.5f + 1.0f) * 256.0f);
  const int p0x = (int)((in[2] * 0.5f + 1.0f) * 256.0f);
  const int p1y = (int)((in[7] * 0.5f + 1.0f) * 256.0f);
  const int p1x = (int)((in[8] * 0.5f + 1.0f) * 256.0f);
  const float scale = 1.0f / 4096.0f;

  for (int t = threadIdx.x; t < TOT; t += blockDim.x) {
    const int g = t / 656, tp = t % 656;
    int c, r, k;
    if (tp < 36)        { c = tp / 9;                 r = tp % 9;  k = 3; }
    else if (tp < 136)  { int u = tp - 36;  c = 4  + u / 25; r = u % 25; k = 5; }
    else if (tp < 332)  { int u = tp - 136; c = 8  + u / 49; r = u % 49; k = 7; }
    else                { int u = tp - 332; c = 12 + u / 81; r = u % 81; k = 9; }
    const int ch = g * 16 + c;
    int py = (g ? p1y : p0y) + (r % k) - (k >> 1);
    int px = (g ? p1x : p0x) + (r / k) - (k >> 1);
    py = min(max(py, 4), 252);
    px = min(max(px, 4), 252);
    const float v = feat[(size_t)pid * FSTR + t] * scale;
    atomicAdd(&grid[(((size_t)(b * 256 + py)) * 256 + px) * 32 + ch], v);
  }
}

// ---------------------------------------------------------------------------
// Depthwise 7x7 conv, SAME zero padding, NHWC, weights (7,7,1,C)
// ---------------------------------------------------------------------------
__global__ void dwconv7(const float* __restrict__ x, const float* __restrict__ w,
                        float* __restrict__ y, int Bn, int H, int W, int C) {
  const size_t total = (size_t)Bn * H * W * C;
  size_t i = (size_t)blockIdx.x * blockDim.x + threadIdx.x;
  if (i >= total) return;
  const int c = (int)(i % C);
  size_t r    = i / C;
  const int xx = (int)(r % W); r /= W;
  const int yy = (int)(r % H);
  const int b  = (int)(r / H);
  float s = 0.0f;
  for (int ky = 0; ky < 7; ++ky) {
    const int iy = yy + ky - 3;
    if (iy < 0 || iy >= H) continue;
    for (int kx = 0; kx < 7; ++kx) {
      const int ix = xx + kx - 3;
      if (ix < 0 || ix >= W) continue;
      s += x[(((size_t)(b * H + iy)) * W + ix) * C + c] * w[(ky * 7 + kx) * C + c];
    }
  }
  y[i] = s;
}

// ---------------------------------------------------------------------------
// LayerNorm over channel dim; one wave32 per pixel, butterfly reduction
// ---------------------------------------------------------------------------
__global__ void ln_kernel(const float* __restrict__ x, const float* __restrict__ g,
                          const float* __restrict__ bta, float* __restrict__ y,
                          int npix, int C) {
  const int pix  = blockIdx.x * 8 + (threadIdx.x >> 5);
  const int lane = threadIdx.x & 31;
  if (pix >= npix) return;  // uniform per wave
  const float* row = x + (size_t)pix * C;
  float s = 0.0f, s2 = 0.0f;
  for (int c = lane; c < C; c += 32) { const float v = row[c]; s += v; s2 += v * v; }
  for (int off = 16; off > 0; off >>= 1) {
    s  += __shfl_xor(s, off, 32);
    s2 += __shfl_xor(s2, off, 32);
  }
  const float m   = s / (float)C;
  const float var = s2 / (float)C - m * m;
  const float rs  = rsqrtf(var + 1e-6f);
  float* out = y + (size_t)pix * C;
  for (int c = lane; c < C; c += 32) out[c] = (row[c] - m) * rs * g[c] + bta[c];
}

// ---------------------------------------------------------------------------
// im2col for 2x2 stride-2 conv; col = (ky*2+kx)*C + c (HWIO weight flatten)
// ---------------------------------------------------------------------------
__global__ void im2col2x2(const float* __restrict__ x, float* __restrict__ out,
                          int Bn, int H, int W, int C) {
  const int Ho = H >> 1, Wo = W >> 1;
  const size_t total = (size_t)Bn * Ho * Wo * 4 * C;
  size_t i = (size_t)blockIdx.x * blockDim.x + threadIdx.x;
  if (i >= total) return;
  const int c = (int)(i % C);
  size_t r    = i / C;
  const int kx = (int)(r % 2); r /= 2;
  const int ky = (int)(r % 2); r /= 2;
  const int ox = (int)(r % Wo); r /= Wo;
  const int oy = (int)(r % Ho);
  const int b  = (int)(r / Ho);
  const int iy = oy * 2 + ky, ix = ox * 2 + kx;
  out[i] = x[(((size_t)(b * H + iy)) * W + ix) * C + c];
}

// ---------------------------------------------------------------------------
// Nearest 2x upsample of `up` concatenated with `skip` -> rows of K = C1+C2
// ---------------------------------------------------------------------------
__global__ void upcat(const float* __restrict__ up, const float* __restrict__ skip,
                      float* __restrict__ out, int Bn, int H, int W, int C1, int C2) {
  const int K = C1 + C2;
  const size_t total = (size_t)Bn * H * W * K;
  size_t i = (size_t)blockIdx.x * blockDim.x + threadIdx.x;
  if (i >= total) return;
  const int k = (int)(i % K);
  size_t m    = i / K;
  const int xx = (int)(m % W); size_t r = m / W;
  const int yy = (int)(r % H);
  const int b  = (int)(r / H);
  float v;
  if (k < C1)
    v = up[(((size_t)(b * (H >> 1) + (yy >> 1))) * (W >> 1) + (xx >> 1)) * C1 + k];
  else
    v = skip[(((size_t)(b * H + yy)) * W + xx) * C2 + (k - C1)];
  out[i] = v;
}

// ---------------------------------------------------------------------------
// Host-side plumbing
// ---------------------------------------------------------------------------
struct BlockP {
  const float *dw, *gamma, *ln_b, *ln_g, *pw1_b, *pw1_w, *pw2_b, *pw2_w;
};
// JAX pytree flatten order per block (sorted keys): dw, gamma, ln_b, ln_g,
// pw1{b,w}, pw2{b,w}
static inline BlockP getBlock(void* const* d_in, int base) {
  return BlockP{(const float*)d_in[base + 0], (const float*)d_in[base + 1],
                (const float*)d_in[base + 2], (const float*)d_in[base + 3],
                (const float*)d_in[base + 4], (const float*)d_in[base + 5],
                (const float*)d_in[base + 6], (const float*)d_in[base + 7]};
}

static inline dim3 gemm_grid(int M, int N) {
  return dim3((unsigned)((N + 63) / 64), (unsigned)(M / 128));
}

static void run_convnext(hipStream_t s, float* x, int Bn, int H, int W, int C,
                         const BlockP& p, const _Float16* pw1p, const _Float16* pw2p,
                         float* t1, float* t2, float* hid) {
  const int npix  = Bn * H * W;
  const int total = npix * C;
  dwconv7<<<(total + 255) / 256, 256, 0, s>>>(x, p.dw, t1, Bn, H, W, C);
  ln_kernel<<<(npix + 7) / 8, 256, 0, s>>>(t1, p.ln_g, p.ln_b, t2, npix, C);
  gemm_wmma<1><<<gemm_grid(npix, 4 * C), 256, 0, s>>>(
      t2, pw1p, p.pw1_b, hid, npix, 4 * C, C, 4 * C, nullptr, nullptr);
  gemm_wmma<2><<<gemm_grid(npix, C), 256, 0, s>>>(
      hid, pw2p, p.pw2_b, x, npix, C, 4 * C, C, p.gamma, x);
}

extern "C" void kernel_launch(void* const* d_in, const int* in_sizes, int n_in,
                              void* d_out, int out_size, void* d_ws, size_t ws_size,
                              hipStream_t stream) {
  (void)in_sizes; (void)n_in; (void)out_size; (void)ws_size;
  auto F = [&](int i) { return (const float*)d_in[i]; };

  // ---- parameter index map (JAX pytree order: sorted dict keys) ----
  // 0: inputs
  // 1..48:  down blocks (6 x 8), stage order 0,0,1,1,2,2
  // 49..56: downsample[0..1]: {b, ln_b, ln_g, w}
  // 57..62: mlp[0..2]: {b, w}
  // 63..94: up blocks (4 x 8): up0b0, up0b1, up1b0, up1b1
  // 95..98: upconv[0..1]: {b, w}
  // 99..102: upsample[0..1]: {ln_b, ln_g}
  const float* inputs = F(0);

  // ---- workspace layout (f32 elements) ----
  float* ws = (float*)d_ws;
  size_t o = 0;
  float* FEAT = ws + o; o += (size_t)16384 * 1344;        // padded stride
  float* H1   = ws + o; o += (size_t)16384 * 256;
  float* H2   = ws + o; o += (size_t)16384 * 512;
  float* X0   = ws + o; o += (size_t)4 * 256 * 256 * 32;  // grid / stage0 / skip0
  float* X1   = ws + o; o += (size_t)4 * 128 * 128 * 64;  // stage1 / skip1
  float* X2   = ws + o; o += (size_t)4 * 64 * 64 * 128;   // stage2
  float* T1   = ws + o; o += (size_t)4 * 256 * 256 * 32;
  float* T2   = ws + o; o += (size_t)4 * 256 * 256 * 32;
  float* HID  = ws + o; o += (size_t)262144 * 128;        // pw1 hidden (max)
  float* CAT  = ws + o; o += (size_t)262144 * 96;         // im2col / concat (max)
  float* UP   = ws + o; o += (size_t)4 * 256 * 256 * 32;  // up-path activations
  float* INP  = ws + o; o += (size_t)16384 * 32;          // padded MLP input
  _Float16* hcur = (_Float16*)(ws + o);                   // packed-weight region

  // ---- pack all GEMM weights to padded/transposed f16 ----
  auto packB = [&](int widx, int K, int N) -> const _Float16* {
    const int Kp = (K + 31) & ~31, Np = (N + 63) & ~63;
    _Float16* dst = hcur; hcur += (size_t)Np * Kp;
    const size_t tot = (size_t)Np * Kp;
    pack_b_f16<<<(unsigned)((tot + 255) / 256), 256, 0, stream>>>(
        F(widx), dst, K, N, Kp, Np);
    return dst;
  };
  const _Float16* Wm1 = packB(58, 12, 256);
  const _Float16* Wm2 = packB(60, 256, 512);
  const _Float16* Wm3 = packB(62, 512, 1312);
  const int dbase[6] = {1, 9, 17, 25, 33, 41};
  const int dC[6]    = {32, 32, 64, 64, 128, 128};
  const int ubase[4] = {63, 71, 79, 87};
  const int uC[4]    = {64, 64, 32, 32};
  const _Float16 *dpw1[6], *dpw2[6], *upw1[4], *upw2[4];
  for (int j = 0; j < 6; ++j) {
    dpw1[j] = packB(dbase[j] + 5, dC[j], 4 * dC[j]);
    dpw2[j] = packB(dbase[j] + 7, 4 * dC[j], dC[j]);
  }
  for (int j = 0; j < 4; ++j) {
    upw1[j] = packB(ubase[j] + 5, uC[j], 4 * uC[j]);
    upw2[j] = packB(ubase[j] + 7, 4 * uC[j], uC[j]);
  }
  const _Float16* Wd0 = packB(52, 128, 64);
  const _Float16* Wd1 = packB(56, 256, 128);
  const _Float16* Wu0 = packB(96, 192, 64);
  const _Float16* Wu1 = packB(98, 96, 32);

  // ================= 1) MLP point encoder =================
  const int M = 16384;  // 4 * 4096 points
  pad_in_kernel<<<(M * 32 + 255) / 256, 256, 0, stream>>>(inputs, INP);
  gemm_wmma<1><<<gemm_grid(M, 256), 256, 0, stream>>>(
      INP, Wm1, F(57), H1, M, 256, 32, 256, nullptr, nullptr);
  gemm_wmma<1><<<gemm_grid(M, 512), 256, 0, stream>>>(
      H1, Wm2, F(59), H2, M, 512, 256, 512, nullptr, nullptr);
  gemm_wmma<0><<<gemm_grid(M, 1312), 256, 0, stream>>>(
      H2, Wm3, F(61), FEAT, M, 1312, 512, 1344, nullptr, nullptr);

  // ================= 2) Scatter into grid =================
  {
    const size_t ngrid = (size_t)4 * 256 * 256 * 32;
    zero_kernel<<<(unsigned)((ngrid + 255) / 256), 256, 0, stream>>>(X0, ngrid);
    scatter_kernel<<<16384, 256, 0, stream>>>(inputs, FEAT, X0);
  }

  // ================= 3) ConvNeXt U-Net =================
  // down stage 0 (32ch @ 256^2)
  run_convnext(stream, X0, 4, 256, 256, 32, getBlock(d_in, 1), dpw1[0], dpw2[0], T1, T2, HID);
  run_convnext(stream, X0, 4, 256, 256, 32, getBlock(d_in, 9), dpw1[1], dpw2[1], T1, T2, HID);
  // downsample 0: LN + 2x2 s2 conv 32->64
  {
    const int npix = 4 * 256 * 256;
    ln_kernel<<<(npix + 7) / 8, 256, 0, stream>>>(X0, F(51), F(50), T1, npix, 32);
    const size_t tot = (size_t)4 * 128 * 128 * 128;  // M=65536, K=128
    im2col2x2<<<(unsigned)((tot + 255) / 256), 256, 0, stream>>>(T1, CAT, 4, 256, 256, 32);
    gemm_wmma<0><<<gemm_grid(65536, 64), 256, 0, stream>>>(
        CAT, Wd0, F(49), X1, 65536, 64, 128, 64, nullptr, nullptr);
  }
  // down stage 1 (64ch @ 128^2)
  run_convnext(stream, X1, 4, 128, 128, 64, getBlock(d_in, 17), dpw1[2], dpw2[2], T1, T2, HID);
  run_convnext(stream, X1, 4, 128, 128, 64, getBlock(d_in, 25), dpw1[3], dpw2[3], T1, T2, HID);
  // downsample 1: LN + 2x2 s2 conv 64->128
  {
    const int npix = 4 * 128 * 128;
    ln_kernel<<<(npix + 7) / 8, 256, 0, stream>>>(X1, F(55), F(54), T1, npix, 64);
    const size_t tot = (size_t)4 * 64 * 64 * 256;  // M=16384, K=256
    im2col2x2<<<(unsigned)((tot + 255) / 256), 256, 0, stream>>>(T1, CAT, 4, 128, 128, 64);
    gemm_wmma<0><<<gemm_grid(16384, 128), 256, 0, stream>>>(
        CAT, Wd1, F(53), X2, 16384, 128, 256, 128, nullptr, nullptr);
  }
  // down stage 2 (128ch @ 64^2)
  run_convnext(stream, X2, 4, 64, 64, 128, getBlock(d_in, 33), dpw1[4], dpw2[4], T1, T2, HID);
  run_convnext(stream, X2, 4, 64, 64, 128, getBlock(d_in, 41), dpw1[5], dpw2[5], T1, T2, HID);

  // up stage 0: LN, 2x upsample, concat skip1 (64ch), 1x1 conv 192->64
  {
    const int npix = 4 * 64 * 64;
    ln_kernel<<<(npix + 7) / 8, 256, 0, stream>>>(X2, F(100), F(99), T1, npix, 128);
    const size_t tot = (size_t)4 * 128 * 128 * 192;  // M=65536, K=192
    upcat<<<(unsigned)((tot + 255) / 256), 256, 0, stream>>>(T1, X1, CAT, 4, 128, 128, 128, 64);
    gemm_wmma<0><<<gemm_grid(65536, 64), 256, 0, stream>>>(
        CAT, Wu0, F(95), UP, 65536, 64, 192, 64, nullptr, nullptr);
  }
  run_convnext(stream, UP, 4, 128, 128, 64, getBlock(d_in, 63), upw1[0], upw2[0], T1, T2, HID);
  run_convnext(stream, UP, 4, 128, 128, 64, getBlock(d_in, 71), upw1[1], upw2[1], T1, T2, HID);

  // up stage 1: LN, 2x upsample, concat skip0 (32ch), 1x1 conv 96->32 -> d_out
  float* OUT = (float*)d_out;  // (4,256,256,32) f32
  {
    const int npix = 4 * 128 * 128;
    ln_kernel<<<(npix + 7) / 8, 256, 0, stream>>>(UP, F(102), F(101), T1, npix, 64);
    const size_t tot = (size_t)4 * 256 * 256 * 96;  // M=262144, K=96
    upcat<<<(unsigned)((tot + 255) / 256), 256, 0, stream>>>(T1, X0, CAT, 4, 256, 256, 64, 32);
    gemm_wmma<0><<<gemm_grid(262144, 32), 256, 0, stream>>>(
        CAT, Wu1, F(97), OUT, 262144, 32, 96, 32, nullptr, nullptr);
  }
  run_convnext(stream, OUT, 4, 256, 256, 32, getBlock(d_in, 79), upw1[2], upw2[2], T1, T2, HID);
  run_convnext(stream, OUT, 4, 256, 256, 32, getBlock(d_in, 87), upw1[3], upw2[3], T1, T2, HID);
}
